// LSTM_24077586661464
// MI455X (gfx1250) — matrix operand order repaired and checked by reading, not compiled, not run on hardware
//
#include <hip/hip_runtime.h>

typedef _Float16 v16h __attribute__((ext_vector_type(16)));
typedef _Float16 v8h  __attribute__((ext_vector_type(8)));
typedef _Float16 v2h  __attribute__((ext_vector_type(2)));
typedef float    v8f  __attribute__((ext_vector_type(8)));
typedef float    v4f  __attribute__((ext_vector_type(4)));
typedef float    v2f  __attribute__((ext_vector_type(2)));

#define E_ 32
#define H_ 128
#define V_ 32000
#define B_ 32
#define T_ 128

// workspace byte offsets
#define WS_XCAT   0u          // T*B*512 f32  = 8,388,608 B
#define WS_UCAT16 8388608u    // 512*128 f16  =   131,072 B
#define WS_VW16   8519680u    // 32000*128 f16 = 8,192,000 B
#define WS_H16    16711680u   // B*T*H f16    = 1,048,576 B  (total ~17.8 MB)

// LDS layout for lstm_seq (bytes)
#define SG_STRIDE 516                         // floats; 8*516 % 64 == 32 -> no bank clash
#define LDS_SH 0
#define LDS_SC 4096
#define LDS_SG 12288                          // 16*516*4 = 33024
#define LDS_SU 45312                          // 512*128*2 = 131072
#define LDS_TOTAL (45312 + 131072)            // 176384 of the WGP's 320 KB

__device__ __forceinline__ float fast_exp(float x) { return __expf(x); }
__device__ __forceinline__ float fast_rcp(float x) { return __builtin_amdgcn_rcpf(x); }

// ---------------------------------------------------------------------------
// Phase 1: xcat[t][b][g*128+j] = x[b,t,:]·Wg[j,:] + bWg[j] + bUg[j]
// ---------------------------------------------------------------------------
__global__ __launch_bounds__(256) void xcat_kernel(
    const float* __restrict__ x,
    const float* __restrict__ Wf, const float* __restrict__ Wi,
    const float* __restrict__ Wc, const float* __restrict__ Wo,
    const float* __restrict__ bWf, const float* __restrict__ bWi,
    const float* __restrict__ bWc, const float* __restrict__ bWo,
    const float* __restrict__ bUf, const float* __restrict__ bUi,
    const float* __restrict__ bUc, const float* __restrict__ bUo,
    float* __restrict__ xcat)
{
    int idx = blockIdx.x * 256 + threadIdx.x;       // [0, T*B*512)
    int n = idx & 511;
    int b = (idx >> 9) & 31;
    int t = idx >> 14;
    int g = n >> 7, j = n & 127;
    const float* W  = (g == 0) ? Wf  : (g == 1) ? Wi  : (g == 2) ? Wc  : Wo;
    const float* bW = (g == 0) ? bWf : (g == 1) ? bWi : (g == 2) ? bWc : bWo;
    const float* bU = (g == 0) ? bUf : (g == 1) ? bUi : (g == 2) ? bUc : bUo;
    const v4f* xr = (const v4f*)(x + ((size_t)b * T_ + t) * E_);
    const v4f* wr = (const v4f*)(W + (size_t)j * E_);
    float s = 0.f;
#pragma unroll
    for (int q = 0; q < 8; ++q) {
        v4f a = xr[q], w = wr[q];
        s += a.x * w.x + a.y * w.y + a.z * w.z + a.w * w.w;
    }
    xcat[idx] = s + bW[j] + bU[j];
}

// ---------------------------------------------------------------------------
// Weight conversion to f16.  u16[n][k] = Ug[j][k]  (B-matrix = U^T, concat f,i,c,o)
// ---------------------------------------------------------------------------
__global__ __launch_bounds__(256) void cvt_u(
    const float* __restrict__ Uf, const float* __restrict__ Ui,
    const float* __restrict__ Uc, const float* __restrict__ Uo,
    _Float16* __restrict__ u16)
{
    int idx = blockIdx.x * 256 + threadIdx.x;       // [0, 512*128)
    int n = idx >> 7, k = idx & 127;
    int g = n >> 7, j = n & 127;
    const float* U = (g == 0) ? Uf : (g == 1) ? Ui : (g == 2) ? Uc : Uo;
    u16[idx] = (_Float16)U[j * 128 + k];
}

// 8 floats -> 8 halfs per thread (16B stores)
__global__ __launch_bounds__(256) void cvt_v(const float* __restrict__ Vw,
                                             _Float16* __restrict__ vw16)
{
    size_t base = ((size_t)blockIdx.x * 256 + threadIdx.x) * 8;   // [0, V*H)
    v4f a = *(const v4f*)(Vw + base);
    v4f b = *(const v4f*)(Vw + base + 4);
    v8h o;
#pragma unroll
    for (int e = 0; e < 4; ++e) { o[e] = (_Float16)a[e]; o[4 + e] = (_Float16)b[e]; }
    *(v8h*)(vw16 + base) = o;
}

// ---------------------------------------------------------------------------
// Phase 2: serial LSTM recurrence.  2 blocks x 16 batches, 32 waves each.
// Gates GEMM [16,128]x[128,512] per step via v_wmma_f32_16x16x32_f16.
// ---------------------------------------------------------------------------
__global__ __launch_bounds__(1024) void lstm_seq(
    const float* __restrict__ xcat, const _Float16* __restrict__ u16g,
    _Float16* __restrict__ h16all, float* __restrict__ hlast)
{
    extern __shared__ char smem[];
    _Float16* sH = (_Float16*)(smem + LDS_SH);      // [16][128] f16
    float*    sC = (float*)(smem + LDS_SC);         // [16][128] f32
    float*    sG = (float*)(smem + LDS_SG);         // [16][SG_STRIDE] f32
    _Float16* sU = (_Float16*)(smem + LDS_SU);      // [512][128] f16

    const int tid  = threadIdx.x;
    const int wave = tid >> 5;
    const int lane = tid & 31;
    const int hid  = lane >> 4;
    const int lm   = lane & 15;
    const int b0   = blockIdx.x * 16;

    // one-time preload of U^T (f16, 128 KB) into LDS
    {
        const uint32_t* src = (const uint32_t*)u16g;
        uint32_t*       dst = (uint32_t*)sU;
        for (int i = tid; i < (512 * 128) / 2; i += 1024) dst[i] = src[i];
    }
    if (tid < 2048) { sH[tid] = (_Float16)0.f; sC[tid] = 0.f; }
    __syncthreads();

    const int nidx = wave * 16 + lm;          // output column 0..511
    const int g    = wave >> 3;               // gate: 0=f 1=i 2=c~ 3=o
    // act(v) = a * rcp(1 + exp(-a*v)) + b  ==  sigmoid (a=1,b=0) / tanh (a=2,b=-1)
    const float act_a = (g == 2) ? 2.0f : 1.0f;
    const float act_b = (g == 2) ? -1.0f : 0.0f;
    const _Float16* pb_base = sU + nidx * 128 + hid * 16;

    // elementwise phase mapping: each thread owns 2 adjacent j
    const int ebase = tid * 2;                // [0, 2048)
    const int em    = ebase >> 7;
    const int ej    = ebase & 127;

    for (int t = 0; t < T_; ++t) {
        // --- h @ U^T tile for this wave (16x16, K=128 as 4-wmma chain) ---
        v8f acc = {};
#pragma unroll
        for (int kk = 0; kk < 4; ++kk) {
            const _Float16* pa = sH + lm * 128 + kk * 32 + hid * 16;
            v8h alo = *(const v8h*)pa;
            v8h ahi = *(const v8h*)(pa + 8);
            v16h a;
#pragma unroll
            for (int e = 0; e < 8; ++e) { a[e] = alo[e]; a[8 + e] = ahi[e]; }
            v16h b = *(const v16h*)(pb_base + kk * 32);
            acc = __builtin_amdgcn_wmma_f32_16x16x32_f16(
                false, a, false, b, (short)0, acc, false, false);
        }
        // --- add x-projection (+biases folded), branchless activation, to LDS ---
        const float* xrow = xcat + ((size_t)t * 32 + b0) * 512 + nidx;
#pragma unroll
        for (int r = 0; r < 8; ++r) {
            int m   = r + 8 * hid;
            float v = acc[r] + xrow[(size_t)m * 512];
            float rc = fast_rcp(1.0f + fast_exp(-act_a * v));
            sG[m * SG_STRIDE + nidx] = act_a * rc + act_b;
        }
        __syncthreads();
        // --- elementwise cell/hidden update (2 adjacent j per thread) ---
        {
            const float* grow = sG + em * SG_STRIDE + ej;
            v2f f2 = *(const v2f*)(grow);
            v2f i2 = *(const v2f*)(grow + 128);
            v2f t2 = *(const v2f*)(grow + 256);
            v2f o2 = *(const v2f*)(grow + 384);
            v2f c2 = *(const v2f*)(sC + ebase);
            float cn0 = f2.x * c2.x + i2.x * t2.x;
            float cn1 = f2.y * c2.y + i2.y * t2.y;
            float th0 = 2.0f * fast_rcp(1.0f + fast_exp(-2.0f * cn0)) - 1.0f;
            float th1 = 2.0f * fast_rcp(1.0f + fast_exp(-2.0f * cn1)) - 1.0f;
            float hn0 = o2.x * th0;
            float hn1 = o2.y * th1;
            v2f cn; cn.x = cn0; cn.y = cn1;
            *(v2f*)(sC + ebase) = cn;
            v2h hh; hh[0] = (_Float16)hn0; hh[1] = (_Float16)hn1;
            *(v2h*)(sH + ebase) = hh;
            *(v2h*)(h16all + ((size_t)(b0 + em) * T_ + t) * H_ + ej) = hh;
            if (t == T_ - 1) {
                v2f hv; hv.x = hn0; hv.y = hn1;
                *(v2f*)(hlast + (b0 + em) * H_ + ej) = hv;
            }
        }
        __syncthreads();
    }
}

// ---------------------------------------------------------------------------
// Phase 3: out[m][n] = h16all[m,:]·vw16[n,:] + Vb[n]
// M=4096, N=32000, K=128.  8 waves/block; wave = one 16x64 tile.
// ---------------------------------------------------------------------------
__global__ __launch_bounds__(256) void vocab_gemm(
    const _Float16* __restrict__ h16all, const _Float16* __restrict__ vw16,
    const float* __restrict__ vb, float* __restrict__ out)
{
    const int tid  = threadIdx.x;
    const int wave = tid >> 5;
    const int lane = tid & 31;
    const int hid  = lane >> 4;
    const int lm   = lane & 15;
    const int m0   = blockIdx.y * 128 + wave * 16;
    const int n0   = blockIdx.x * 64;

    v16h A[4];
#pragma unroll
    for (int kk = 0; kk < 4; ++kk) {
        const _Float16* pa = h16all + (size_t)(m0 + lm) * H_ + kk * 32 + hid * 16;
        v8h lo = *(const v8h*)pa;
        v8h hi = *(const v8h*)(pa + 8);
#pragma unroll
        for (int e = 0; e < 8; ++e) { A[kk][e] = lo[e]; A[kk][8 + e] = hi[e]; }
    }
#pragma unroll
    for (int ns = 0; ns < 4; ++ns) {
        int n = n0 + ns * 16 + lm;
        const _Float16* pb = vw16 + (size_t)n * H_ + hid * 16;
        if (ns < 3) __builtin_prefetch(pb + H_ * 16, 0, 3);   // next B subtile
        v8f acc = {};
#pragma unroll
        for (int kk = 0; kk < 4; ++kk) {
            v16h b = *(const v16h*)(pb + kk * 32);
            acc = __builtin_amdgcn_wmma_f32_16x16x32_f16(
                false, A[kk], false, b, (short)0, acc, false, false);
        }
        float bias = vb[n];
#pragma unroll
        for (int r = 0; r < 8; ++r) {
            // streaming 524 MB output: non-temporal so Vw16 stays L2-resident
            __builtin_nontemporal_store(acc[r] + bias,
                &out[(size_t)(m0 + r + 8 * hid) * V_ + n]);
        }
    }
}

// ---------------------------------------------------------------------------
extern "C" void kernel_launch(void* const* d_in, const int* in_sizes, int n_in,
                              void* d_out, int out_size, void* d_ws, size_t ws_size,
                              hipStream_t stream)
{
    const float* x   = (const float*)d_in[0];
    const float* Wf  = (const float*)d_in[1];  const float* bWf = (const float*)d_in[2];
    const float* Uf  = (const float*)d_in[3];  const float* bUf = (const float*)d_in[4];
    const float* Wi  = (const float*)d_in[5];  const float* bWi = (const float*)d_in[6];
    const float* Ui  = (const float*)d_in[7];  const float* bUi = (const float*)d_in[8];
    const float* Wc  = (const float*)d_in[9];  const float* bWc = (const float*)d_in[10];
    const float* Uc  = (const float*)d_in[11]; const float* bUc = (const float*)d_in[12];
    const float* Wo  = (const float*)d_in[13]; const float* bWo = (const float*)d_in[14];
    const float* Uo  = (const float*)d_in[15]; const float* bUo = (const float*)d_in[16];
    const float* Vw  = (const float*)d_in[17]; const float* Vb  = (const float*)d_in[18];

    char* ws = (char*)d_ws;
    float*    xcat   = (float*)(ws + WS_XCAT);
    _Float16* u16    = (_Float16*)(ws + WS_UCAT16);
    _Float16* vw16   = (_Float16*)(ws + WS_VW16);
    _Float16* h16all = (_Float16*)(ws + WS_H16);

    float* out   = (float*)d_out;
    float* hlast = out + (size_t)B_ * T_ * V_;   // tuple tail: h_last [B,H]

    xcat_kernel<<<(T_ * B_ * 512) / 256, 256, 0, stream>>>(
        x, Wf, Wi, Wc, Wo, bWf, bWi, bWc, bWo, bUf, bUi, bUc, bUo, xcat);
    cvt_u<<<(512 * 128) / 256, 256, 0, stream>>>(Uf, Ui, Uc, Uo, u16);
    cvt_v<<<(V_ * H_) / (256 * 8), 256, 0, stream>>>(Vw, vw16);

    lstm_seq<<<2, 1024, LDS_TOTAL, stream>>>(xcat, u16, h16all, hlast);

    dim3 grid(V_ / 64, (B_ * T_) / 128);         // 500 x 32 blocks
    vocab_gemm<<<grid, 256, 0, stream>>>(h16all, vw16, Vb, out);
}
